// NEATLayer_31533649887725
// MI455X (gfx1250) — matrix operand order repaired and checked by reading, not compiled, not run on hardware
//
#include <hip/hip_runtime.h>
#include <hip/hip_bf16.h>

// ---------------------------------------------------------------------------
// NEAT 4-layer MLP, batch 65536. Strategy:
//  * The sparse per-node gather uses batch-invariant indices -> densify each
//    layer's weights once into small dense bf16 matrices W[K][Npad] in d_ws
//    (K = 64/192/288/320, Npad = 128/96/32/16), summing duplicate indices.
//  * One fused kernel: each wave owns 16 batch rows; activations live in LDS
//    as bf16 (64 rows x 328 cols, 41KB) across all 4 layers -> HBM traffic is
//    just x (16MB) + out (2.6MB) ~= 0.85us floor at 23.3 TB/s.
//  * Matmul core: v_wmma_f32_16x16x32_bf16, fp32 accumulation, fp32 sigmoid.
// ---------------------------------------------------------------------------

typedef __attribute__((ext_vector_type(16))) __bf16        v16bf;
typedef __attribute__((ext_vector_type(8)))  float         v8f;
typedef __attribute__((ext_vector_type(4)))  unsigned int  v4u;
typedef __attribute__((ext_vector_type(2)))  float         v2f;

#define VSTRIDE        328   // 320 live cols + 8 pad: 656B row stride -> A-frag reads conflict-free
#define ROWS_PER_BLOCK 64
#define THREADS        128   // 4 waves, each wave privately owns 16 rows -> no barriers

__device__ __forceinline__ unsigned short f2bf(float f) {
  unsigned int x = __float_as_uint(f);
  x += 0x7FFFu + ((x >> 16) & 1u);          // round-to-nearest-even
  return (unsigned short)(x >> 16);
}

union AFrag { v16bf v; unsigned int u[8]; };

// One layer: z = vals[:, 0:KT*32] @ W + b ; sigmoid(5z) -> LDS (or d_out if FINAL)
template <int KT, int NT, int NCOLS, bool FINAL>
__device__ __forceinline__ void neat_layer(
    unsigned short (*vals)[VSTRIDE], int lRow0, int lane,
    const unsigned short* __restrict__ W, const float* __restrict__ bias,
    int outBase, float* __restrict__ out, int gRow0)
{
  const int nn   = lane & 15;       // N index within tile (C/D layout)
  const int m0   = lane & 15;       // M row this lane reads for the A matrix
  const int half = lane >> 4;
  const int kb   = half * 8;        // A layout: lanes 0-15 K={0..7,16..23}, 16-31 K={8..15,24..31}
  const unsigned short* arow = vals[lRow0 + m0];
  constexpr int NPAD = NT * 16;

#pragma unroll
  for (int nt = 0; nt < NT; ++nt) {
    const int ncol = nt * 16 + nn;
    const float bv = (ncol < NCOLS) ? bias[ncol] : 0.0f;
    v8f c;
#pragma unroll
    for (int r = 0; r < 8; ++r) c[r] = bv;   // bias broadcast: all 8 M rows share n

#pragma unroll
    for (int kt = 0; kt < KT; ++kt) {
      // A fragment (16x32 bf16) from LDS: 8 x 32-bit reads per lane
      AFrag a;
      const unsigned short* ap = arow + kt * 32 + kb;
#pragma unroll
      for (int v = 0; v < 4; ++v)
        a.u[v] = *(const unsigned int*)(ap + 2 * v);
#pragma unroll
      for (int v = 0; v < 4; ++v)
        a.u[4 + v] = *(const unsigned int*)(ap + 16 + 2 * v);

      // B fragment (32x16 bf16): lane = N column, K order same as A (kb+0..7, kb+16..23)
      AFrag b;
      const unsigned short* wp = W + (size_t)(kt * 32 + kb) * NPAD + nt * 16 + nn;
#pragma unroll
      for (int v = 0; v < 4; ++v)
        b.u[v] = (unsigned int)wp[(2 * v) * NPAD] | ((unsigned int)wp[(2 * v + 1) * NPAD] << 16);
#pragma unroll
      for (int v = 0; v < 4; ++v)
        b.u[4 + v] = (unsigned int)wp[(16 + 2 * v) * NPAD] | ((unsigned int)wp[(17 + 2 * v) * NPAD] << 16);

      c = __builtin_amdgcn_wmma_f32_16x16x32_bf16(
              /*neg_a=*/false, a.v, /*neg_b=*/false, b.v,
              /*c_mod=*/(short)0, c, /*reuse_a=*/false, /*reuse_b=*/false);
    }

    // sigmoid(5z) in fp32, then write out. Masked store only AFTER all WMMAs
    // (EXEC must be all ones during matrix ops).
#pragma unroll
    for (int r = 0; r < 8; ++r) {
      const float s = 1.0f / (1.0f + __expf(-5.0f * c[r]));
      const int m = r + half * 8;   // C/D layout: lanes 0-15 -> M=r, 16-31 -> M=8+r
      if (FINAL) {
        if (nn < NCOLS) out[(size_t)(gRow0 + m) * NCOLS + nn] = s;
      } else {
        vals[lRow0 + m][outBase + ncol] = f2bf(s);
      }
    }
  }
}

__global__ __launch_bounds__(THREADS) void neat_forward(
    const float* __restrict__ x,
    const unsigned short* __restrict__ W0, const unsigned short* __restrict__ W1,
    const unsigned short* __restrict__ W2, const unsigned short* __restrict__ W3,
    const float* __restrict__ b0, const float* __restrict__ b1,
    const float* __restrict__ b2, const float* __restrict__ b3,
    float* __restrict__ out)
{
  __shared__ __align__(16) unsigned short vals[ROWS_PER_BLOCK][VSTRIDE];
  const int lane  = threadIdx.x & 31;
  const int wave  = threadIdx.x >> 5;
  const int lRow0 = wave * 16;
  const int gRow0 = blockIdx.x * ROWS_PER_BLOCK + lRow0;

  // Load this wave's 16 rows x 64 input cols, convert fp32 -> bf16 into LDS.
#pragma unroll
  for (int i = 0; i < 16; ++i) {
    const v2f v = *(const v2f*)(x + (size_t)(gRow0 + i) * 64 + lane * 2);
    const unsigned int p =
        (unsigned int)f2bf(v.x) | ((unsigned int)f2bf(v.y) << 16);
    *(unsigned int*)&vals[lRow0 + i][lane * 2] = p;
  }
  // Wave-private rows -> no barriers anywhere; DScnt keeps LDS ops in order.

  neat_layer<2,  8, 128, false>(vals, lRow0, lane, W0, b0,  64, nullptr, gRow0);
  neat_layer<6,  6,  96, false>(vals, lRow0, lane, W1, b1, 192, nullptr, gRow0);
  neat_layer<9,  2,  32, false>(vals, lRow0, lane, W2, b2, 288, nullptr, gRow0);
  neat_layer<10, 1,  10, true >(vals, lRow0, lane, W3, b3, 320, out,     gRow0);
}

// ---------------------------------------------------------------------------
// Weight densification: scatter w[n,k] into dense bf16 W[src[n,k]][n],
// summing duplicate source indices in fp32. Column n is exclusive to one
// thread -> no atomics needed.
// ---------------------------------------------------------------------------
__global__ void zero_bf16(unsigned short* p, int n) {
  const int i = blockIdx.x * blockDim.x + threadIdx.x;
  if (i < n) p[i] = 0;
}

__global__ void scatter_w(const int* __restrict__ src, const float* __restrict__ w,
                          unsigned short* __restrict__ W, int nNodes, int npad) {
  const int n = blockIdx.x * blockDim.x + threadIdx.x;
  if (n >= nNodes) return;
  for (int k = 0; k < 16; ++k) {
    const int idx = src[n * 16 + k];
    bool dup = false;
    for (int j = 0; j < k; ++j)
      if (src[n * 16 + j] == idx) { dup = true; break; }
    if (dup) continue;                       // already folded into an earlier k
    float s = 0.0f;
    for (int j = k; j < 16; ++j)
      if (src[n * 16 + j] == idx) s += w[n * 16 + j];
    W[(size_t)idx * npad + n] = f2bf(s);
  }
}

extern "C" void kernel_launch(void* const* d_in, const int* in_sizes, int n_in,
                              void* d_out, int out_size, void* d_ws, size_t ws_size,
                              hipStream_t stream) {
  const float* x  = (const float*)d_in[0];
  const int*   s0 = (const int*)d_in[1];
  const float* w0 = (const float*)d_in[2];
  const float* b0 = (const float*)d_in[3];
  const int*   s1 = (const int*)d_in[4];
  const float* w1 = (const float*)d_in[5];
  const float* b1 = (const float*)d_in[6];
  const int*   s2 = (const int*)d_in[7];
  const float* w2 = (const float*)d_in[8];
  const float* b2 = (const float*)d_in[9];
  const int*   s3 = (const int*)d_in[10];
  const float* w3 = (const float*)d_in[11];
  const float* b3 = (const float*)d_in[12];

  // Dense bf16 weight matrices in workspace: 80 KB total.
  unsigned short* W0 = (unsigned short*)d_ws;      // 64  x 128
  unsigned short* W1 = W0 + 64 * 128;              // 192 x 96
  unsigned short* W2 = W1 + 192 * 96;              // 288 x 32
  unsigned short* W3 = W2 + 288 * 32;              // 320 x 16 (cols 10..15 zero)
  const int totalW = 64 * 128 + 192 * 96 + 288 * 32 + 320 * 16;  // 40960 elems

  zero_bf16<<<(totalW + 255) / 256, 256, 0, stream>>>(W0, totalW);
  scatter_w<<<1, 128, 0, stream>>>(s0, w0, W0, 128, 128);
  scatter_w<<<1, 128, 0, stream>>>(s1, w1, W1,  96,  96);
  scatter_w<<<1,  64, 0, stream>>>(s2, w2, W2,  32,  32);
  scatter_w<<<1,  32, 0, stream>>>(s3, w3, W3,  10,  16);

  neat_forward<<<65536 / ROWS_PER_BLOCK, THREADS, 0, stream>>>(
      x, W0, W1, W2, W3, b0, b1, b2, b3, (float*)d_out);
}
